// Policy_31842887533163
// MI455X (gfx1250) — matrix operand order repaired and checked by reading, not compiled, not run on hardware
//
#include <hip/hip_runtime.h>
#include <hip/hip_bf16.h>
#include <math.h>

// ---------------------------------------------------------------------------
// 2-layer GCN for MI455X (gfx1250, wave32, WMMA).
// GEMM: split-bf16 (hi/lo) with V_WMMA_F32_16X16X32_BF16, f32 accumulate:
//   x*w ~= hi(x)*hi(w) + hi(x)*lo(w) + lo(x)*hi(w)   (near-fp32 accuracy)
// Weights pre-split into bf16 hi/lo arrays once (L2-resident, 512KB each).
// Wave tile 16M x 128N: 24 WMMA per k-step against 20 vmem + 24 cvt VALU.
// ---------------------------------------------------------------------------

typedef __bf16  bf16_t;
typedef __attribute__((ext_vector_type(16))) __bf16 v16bf;
typedef __attribute__((ext_vector_type(8)))  float  v8f;

#define IN_DIM 512   // K
#define OUT_DIM 512  // N (output columns)

// ---------------------------------------------------------------------------
// Kernel 0a: rdeg[i] = 1/sqrt(1 + #neighbors with index >= 0)
// ---------------------------------------------------------------------------
__global__ void gcn_rdeg_kernel(const long long* __restrict__ edge,
                                float* __restrict__ rdeg, int n) {
    int i = blockIdx.x * blockDim.x + threadIdx.x;
    if (i < n) {
        const long long* e = edge + (size_t)i * 3;
        int d = 1 + (e[0] >= 0 ? 1 : 0) + (e[1] >= 0 ? 1 : 0) + (e[2] >= 0 ? 1 : 0);
        rdeg[i] = rsqrtf((float)d);
    }
}

// ---------------------------------------------------------------------------
// Kernel 0b: split-pack an f32 weight matrix into bf16 hi/lo (same layout).
// ---------------------------------------------------------------------------
__global__ void gcn_pack_w_kernel(const float* __restrict__ W,
                                  bf16_t* __restrict__ Whi,
                                  bf16_t* __restrict__ Wlo, int n) {
    int i = blockIdx.x * blockDim.x + threadIdx.x;
    if (i < n) {
        float w = W[i];
        bf16_t h = (bf16_t)w;
        Whi[i] = h;
        Wlo[i] = (bf16_t)(w - (float)h);
    }
}

// ---------------------------------------------------------------------------
// Kernel 1: H = (X @ W^T + b) * rdeg[row]
// Grid: one block per 16-row M tile. Block = 128 threads = 4 waves.
// Wave w owns output columns [w*128, w*128+128): 8 WMMA N-tiles, K step 32.
// All 4 waves stream the same 16 X rows -> WGP$/L2 reuse; X hits HBM once.
// ---------------------------------------------------------------------------
__global__ __launch_bounds__(128) void gcn_gemm_kernel(
    const float*  __restrict__ X,     // [M, 512] f32
    const bf16_t* __restrict__ Whi,   // [512, 512] bf16 hi (row-major, [out][in])
    const bf16_t* __restrict__ Wlo,   // [512, 512] bf16 lo
    const float*  __restrict__ bias,  // [512]
    const float*  __restrict__ rdeg,  // [M]
    float* __restrict__ H,            // [M, 512]
    int M)
{
    const int lane = threadIdx.x & 31;
    const int wave = threadIdx.x >> 5;   // 0..3
    const int half = lane >> 4;          // 0: lanes 0-15, 1: lanes 16-31
    const int l16  = lane & 15;

    const int m0 = blockIdx.x * 16;
    const int n0 = wave * 128;

    // Clamp A-row for the (possible) M tail; stores are guarded below.
    int arow_idx = m0 + l16;
    if (arow_idx >= M) arow_idx = M - 1;
    const float* __restrict__ arow = X + (size_t)arow_idx * IN_DIM;

    v8f acc[8];
    #pragma unroll
    for (int t = 0; t < 8; ++t) acc[t] = (v8f){0.f,0.f,0.f,0.f,0.f,0.f,0.f,0.f};

    #pragma unroll 1
    for (int k0 = 0; k0 < IN_DIM; k0 += 32) {
        // ---- A fragment (16x32 bf16), contiguous-k layout (matches B): lane holds
        // K = [k0+16*half .. +15] of row m0+l16.
        v16bf a_hi, a_lo;
        {
            const float4* p0 = (const float4*)(arow + k0 + 16 * half);
            const float4* p1 = (const float4*)(arow + k0 + 16 * half + 8);
            float av[16];
            float4 q;
            q = p0[0]; av[0]=q.x;  av[1]=q.y;  av[2]=q.z;  av[3]=q.w;
            q = p0[1]; av[4]=q.x;  av[5]=q.y;  av[6]=q.z;  av[7]=q.w;
            q = p1[0]; av[8]=q.x;  av[9]=q.y;  av[10]=q.z; av[11]=q.w;
            q = p1[1]; av[12]=q.x; av[13]=q.y; av[14]=q.z; av[15]=q.w;
            #pragma unroll
            for (int i = 0; i < 16; ++i) {
                bf16_t h = (bf16_t)av[i];
                a_hi[i] = h;
                a_lo[i] = (bf16_t)(av[i] - (float)h);
            }
        }
        // Speculative prefetch of next iteration's A chunk (safe at tail).
        __builtin_prefetch(arow + k0 + 32 + 8 * half, 0, 3);

        #pragma unroll
        for (int t = 0; t < 8; ++t) {
            // ---- B fragment (32x16 bf16): lane holds column n = n0+16t+l16,
            // K = k0 + 16*half + [0..15] -- contiguous 32B in prepacked array.
            const size_t boff =
                (size_t)(n0 + 16 * t + l16) * IN_DIM + (size_t)(k0 + 16 * half);
            v16bf b_hi = *(const v16bf*)(Whi + boff);
            v16bf b_lo = *(const v16bf*)(Wlo + boff);

            // Split-product accumulation (hi*hi + hi*lo + lo*hi), f32 acc.
            acc[t] = __builtin_amdgcn_wmma_f32_16x16x32_bf16(
                false, a_hi, false, b_hi, (short)0, acc[t], false, false);
            acc[t] = __builtin_amdgcn_wmma_f32_16x16x32_bf16(
                false, a_hi, false, b_lo, (short)0, acc[t], false, false);
            acc[t] = __builtin_amdgcn_wmma_f32_16x16x32_bf16(
                false, a_lo, false, b_hi, (short)0, acc[t], false, false);
        }
    }

    // ---- Epilogue: D layout row = m0 + vi + 8*half, col = n0 + 16*t + l16.
    #pragma unroll
    for (int vi = 0; vi < 8; ++vi) {
        int row = m0 + vi + 8 * half;
        if (row < M) {
            float r = rdeg[row];
            float* hrow = H + (size_t)row * OUT_DIM;
            #pragma unroll
            for (int t = 0; t < 8; ++t) {
                int col = n0 + 16 * t + l16;
                hrow[col] = (acc[t][vi] + bias[col]) * r;
            }
        }
    }
}

// ---------------------------------------------------------------------------
// Kernel 2: out[i] = ELU( (H[i] + sum_{e>=0} H[e]) * rdeg[i] )
// One block per node, 128 threads, float4 per thread.
// ---------------------------------------------------------------------------
__global__ __launch_bounds__(128) void gcn_agg_elu_kernel(
    const float* __restrict__ H,        // [N, 512] pre-aggregation
    const long long* __restrict__ edge, // [N, 3]
    const float* __restrict__ rdeg,     // [N]
    float* __restrict__ out, int n)
{
    const int i = blockIdx.x;
    const int c = threadIdx.x;  // float4 index, 0..127
    const long long e0 = edge[(size_t)i * 3 + 0];
    const long long e1 = edge[(size_t)i * 3 + 1];
    const long long e2 = edge[(size_t)i * 3 + 2];
    const float r = rdeg[i];

    float4 s = ((const float4*)(H + (size_t)i * OUT_DIM))[c];
    if (e0 >= 0) {
        float4 v = ((const float4*)(H + (size_t)e0 * OUT_DIM))[c];
        s.x += v.x; s.y += v.y; s.z += v.z; s.w += v.w;
    }
    if (e1 >= 0) {
        float4 v = ((const float4*)(H + (size_t)e1 * OUT_DIM))[c];
        s.x += v.x; s.y += v.y; s.z += v.z; s.w += v.w;
    }
    if (e2 >= 0) {
        float4 v = ((const float4*)(H + (size_t)e2 * OUT_DIM))[c];
        s.x += v.x; s.y += v.y; s.z += v.z; s.w += v.w;
    }
    s.x *= r; s.y *= r; s.z *= r; s.w *= r;
    // ELU (alpha = 1)
    s.x = (s.x > 0.f) ? s.x : (__expf(s.x) - 1.f);
    s.y = (s.y > 0.f) ? s.y : (__expf(s.y) - 1.f);
    s.z = (s.z > 0.f) ? s.z : (__expf(s.z) - 1.f);
    s.w = (s.w > 0.f) ? s.w : (__expf(s.w) - 1.f);
    ((float4*)(out + (size_t)i * OUT_DIM))[c] = s;
}

// ---------------------------------------------------------------------------
// Launch: rdeg + pack(W1,W2) -> gemm1 -> agg1 -> gemm2 -> agg2 -> d_out
// ws layout: [rdeg (256B aligned)] [W1hi][W1lo][W2hi][W2lo] [bufA][bufB]
// ---------------------------------------------------------------------------
extern "C" void kernel_launch(void* const* d_in, const int* in_sizes, int n_in,
                              void* d_out, int out_size, void* d_ws, size_t ws_size,
                              hipStream_t stream) {
    const float*      x    = (const float*)d_in[0];
    const long long*  edge = (const long long*)d_in[1];
    const float*      W1   = (const float*)d_in[2];
    const float*      b1   = (const float*)d_in[3];
    const float*      W2   = (const float*)d_in[4];
    const float*      b2   = (const float*)d_in[5];
    float*            out  = (float*)d_out;

    const int N = in_sizes[0] / IN_DIM;
    const int WELEMS = OUT_DIM * IN_DIM;  // 512*512

    char* ws = (char*)d_ws;
    size_t off = 0;
    float* rdeg = (float*)(ws + off);
    off += (((size_t)N * sizeof(float)) + 255) & ~(size_t)255;
    bf16_t* W1hi = (bf16_t*)(ws + off); off += (size_t)WELEMS * sizeof(bf16_t);
    bf16_t* W1lo = (bf16_t*)(ws + off); off += (size_t)WELEMS * sizeof(bf16_t);
    bf16_t* W2hi = (bf16_t*)(ws + off); off += (size_t)WELEMS * sizeof(bf16_t);
    bf16_t* W2lo = (bf16_t*)(ws + off); off += (size_t)WELEMS * sizeof(bf16_t);
    off = (off + 255) & ~(size_t)255;
    float* bufA = (float*)(ws + off); off += (size_t)N * OUT_DIM * sizeof(float);
    float* bufB = (float*)(ws + off);

    const int mtiles = (N + 15) / 16;

    gcn_rdeg_kernel<<<(N + 255) / 256, 256, 0, stream>>>(edge, rdeg, N);
    gcn_pack_w_kernel<<<(WELEMS + 255) / 256, 256, 0, stream>>>(W1, W1hi, W1lo, WELEMS);
    gcn_pack_w_kernel<<<(WELEMS + 255) / 256, 256, 0, stream>>>(W2, W2hi, W2lo, WELEMS);

    // Layer 1
    gcn_gemm_kernel<<<mtiles, 128, 0, stream>>>(x, W1hi, W1lo, b1, rdeg, bufA, N);
    gcn_agg_elu_kernel<<<N, 128, 0, stream>>>(bufA, edge, rdeg, bufB, N);

    // Layer 2
    gcn_gemm_kernel<<<mtiles, 128, 0, stream>>>(bufB, W2hi, W2lo, b2, rdeg, bufA, N);
    gcn_agg_elu_kernel<<<N, 128, 0, stream>>>(bufA, edge, rdeg, out, N);
}
